// MultiHeadAttention_7421703487858
// MI455X (gfx1250) — matrix operand order repaired and checked by reading, not compiled, not run on hardware
//
#include <hip/hip_runtime.h>
#include <hip/hip_bf16.h>

typedef __bf16 bf16_t;
typedef __attribute__((ext_vector_type(4)))  __bf16 v4bf;
typedef __attribute__((ext_vector_type(8)))  __bf16 v8bf;
typedef __attribute__((ext_vector_type(16))) __bf16 v16bf;
typedef __attribute__((ext_vector_type(8)))  float  v8f;
typedef __attribute__((ext_vector_type(4)))  int    v4i;

#define B_  4
#define S_  2048
#define D_  1024
#define H_  16
#define HD_ 64
#define M_  (B_ * S_)

// Padded LDS row stride (elements): 40*2B = 80B -> 20-bank step, 16 lanes hit
// 16 distinct banks on fragment loads (vs 4 with a 64B stride).
#define LDSTRIDE 40

#if defined(__has_builtin)
#if __has_builtin(__builtin_amdgcn_global_load_async_to_lds_b128) && \
    __has_builtin(__builtin_amdgcn_s_wait_asynccnt)
#define USE_ASYNC_LDS 1
#endif
#endif
#ifndef USE_ASYNC_LDS
#define USE_ASYNC_LDS 0
#endif

union BF16x16 { v16bf v; v8bf h[2]; };

static __device__ __forceinline__ v8f wmma_bf16(v16bf a, v16bf b, v8f c) {
  return __builtin_amdgcn_wmma_f32_16x16x32_bf16(false, a, false, b, (short)0, c, false, false);
}

// Copy 32B (16 bf16) global -> LDS; async (ASYNCcnt-tracked) when available.
static __device__ __forceinline__ void copy32_g2l(const bf16_t* g, bf16_t* l) {
#if USE_ASYNC_LDS
  __builtin_amdgcn_global_load_async_to_lds_b128(
      (__attribute__((address_space(1))) v4i*)g,
      (__attribute__((address_space(3))) v4i*)l, 0, 0);
  __builtin_amdgcn_global_load_async_to_lds_b128(
      (__attribute__((address_space(1))) v4i*)(g + 8),
      (__attribute__((address_space(3))) v4i*)(l + 8), 0, 0);
#else
  ((v8bf*)l)[0] = ((const v8bf*)g)[0];
  ((v8bf*)l)[1] = ((const v8bf*)g)[1];
#endif
}

// ---------------------------------------------------------------------------
// fp32 -> bf16 (x)
// ---------------------------------------------------------------------------
__global__ __launch_bounds__(256)
void cvt_f32_bf16(const float* __restrict__ in, bf16_t* __restrict__ out, int n) {
  const int i = (blockIdx.x * 256 + threadIdx.x) * 4;
  if (i + 3 < n) {
    const float4 f = *(const float4*)(in + i);
    v4bf o;
    o.x = (bf16_t)f.x; o.y = (bf16_t)f.y; o.z = (bf16_t)f.z; o.w = (bf16_t)f.w;
    *(v4bf*)(out + i) = o;
  } else if (i < n) {
    for (int j = i; j < n; ++j) out[j] = (bf16_t)in[j];
  }
}

// ---------------------------------------------------------------------------
// fp32 [K][N] -> bf16 transposed [N][K] (weights; once per call, so the GEMM
// hot loop never transposes). 64x64 tiles via LDS, coalesced in and out.
// ---------------------------------------------------------------------------
__global__ __launch_bounds__(256)
void cvt_transpose_f32_bf16(const float* __restrict__ in, bf16_t* __restrict__ out,
                            int K, int N) {
  __shared__ alignas(16) bf16_t tile[64][72];  // padded rows
  const int tid = threadIdx.x;
  const int k0 = blockIdx.y * 64, n0 = blockIdx.x * 64;
  const int rr = tid >> 4;          // 0..15
  const int cc = (tid & 15) * 4;    // 0..60
  #pragma unroll
  for (int i = 0; i < 4; ++i) {
    const int k = rr + i * 16;
    const float4 f = *(const float4*)(in + (size_t)(k0 + k) * N + n0 + cc);
    tile[cc + 0][k] = (bf16_t)f.x;
    tile[cc + 1][k] = (bf16_t)f.y;
    tile[cc + 2][k] = (bf16_t)f.z;
    tile[cc + 3][k] = (bf16_t)f.w;
  }
  __syncthreads();
  const int n = tid >> 2;           // 0..63
  const int kc = (tid & 3) * 16;    // 0..48
  bf16_t* dst = out + (size_t)(n0 + n) * K + k0 + kc;
  ((v8bf*)dst)[0] = *(const v8bf*)&tile[n][kc];
  ((v8bf*)dst)[1] = *(const v8bf*)&tile[n][kc + 8];
}

// ---------------------------------------------------------------------------
// bf16 GEMM, f32 accumulate. C = A[M,K] @ Bt[N,K]^T (B pre-transposed).
// 256 threads = 8 waves; tile 128x128; BK=32; wave = 2x4 WMMA 16x16x32 tiles.
// MODE 0: bf16 out [B,H,S,HD] (Q,K)   MODE 2: bf16 out [B,H,HD,S] (V^T)
// MODE 1: fp32 out + bias [M,N] (final projection)
// ---------------------------------------------------------------------------
template<int MODE>
__global__ __launch_bounds__(256)
void gemm_bf16_wmma(const bf16_t* __restrict__ A, const bf16_t* __restrict__ Bt,
                    const float* __restrict__ bias, void* __restrict__ Cout) {
  __shared__ alignas(16) bf16_t lA[128 * LDSTRIDE];
  __shared__ alignas(16) bf16_t lB[128 * LDSTRIDE];

  const int K = D_, N = D_;
  const int tid = threadIdx.x;
  const int blockRow = blockIdx.y * 128;
  const int blockCol = blockIdx.x * 128;
  const int srow = tid >> 1, sk = (tid & 1) * 16;   // staging: 32B per thread
  const int lane = tid & 31, wave = tid >> 5;
  const int wm = wave >> 1, wn = wave & 1;
  const int lg = lane >> 4, ln = lane & 15;

  const bf16_t* gA = A  + (size_t)(blockRow + srow) * K + sk;
  const bf16_t* gB = Bt + (size_t)(blockCol + srow) * K + sk;
  bf16_t* sA = &lA[srow * LDSTRIDE + sk];
  bf16_t* sB = &lB[srow * LDSTRIDE + sk];

  v8f acc[2][4] = {};

  for (int k0 = 0; k0 < K; k0 += 32) {
    __syncthreads();
    copy32_g2l(gA + k0, sA);
    copy32_g2l(gB + k0, sB);
#if USE_ASYNC_LDS
    __builtin_amdgcn_s_wait_asynccnt(0);
#endif
    __syncthreads();

    // A fragment: lane (g,m): elem e -> k = g*16 + e (contiguous)
    v16bf af[2];
    #pragma unroll
    for (int mt = 0; mt < 2; ++mt) {
      BF16x16 u;
      const bf16_t* p = &lA[(wm * 32 + mt * 16 + ln) * LDSTRIDE];
      u.h[0] = *(const v8bf*)(p + lg * 16);
      u.h[1] = *(const v8bf*)(p + lg * 16 + 8);
      af[mt] = u.v;
    }
    // B fragment: lane (g,n): elem e -> k = g*16 + e (contiguous)
    v16bf bfr[4];
    #pragma unroll
    for (int nt = 0; nt < 4; ++nt) {
      BF16x16 u;
      const bf16_t* p = &lB[(wn * 64 + nt * 16 + ln) * LDSTRIDE + lg * 16];
      u.h[0] = *(const v8bf*)(p);
      u.h[1] = *(const v8bf*)(p + 8);
      bfr[nt] = u.v;
    }
    #pragma unroll
    for (int mt = 0; mt < 2; ++mt)
      #pragma unroll
      for (int nt = 0; nt < 4; ++nt)
        acc[mt][nt] = wmma_bf16(af[mt], bfr[nt], acc[mt][nt]);
  }

  // epilogue: C layout lane l, vgpr r -> row = (l>>4)*8 + r, col = l&15
  #pragma unroll
  for (int mt = 0; mt < 2; ++mt) {
    #pragma unroll
    for (int nt = 0; nt < 4; ++nt) {
      #pragma unroll
      for (int r = 0; r < 8; ++r) {
        const int row = blockRow + wm * 32 + mt * 16 + lg * 8 + r;
        const int col = blockCol + wn * 64 + nt * 16 + ln;
        const float v = acc[mt][nt][r];
        if (MODE == 1) {
          ((float*)Cout)[(size_t)row * N + col] = v + bias[col];
        } else {
          const int b = row >> 11, s = row & (S_ - 1);
          const int h = col >> 6,  hd = col & (HD_ - 1);
          size_t idx;
          if (MODE == 0) idx = ((size_t)(b * H_ + h) * S_ + s) * HD_ + hd;
          else           idx = ((size_t)(b * H_ + h) * HD_ + hd) * S_ + s;
          ((bf16_t*)Cout)[idx] = (bf16_t)v;
        }
      }
    }
  }
}

// ---------------------------------------------------------------------------
// Causal flash attention, bf16 WMMA, online softmax. Scores never hit HBM.
// Grid (S/64, B*H), block 128 (4 waves); wave = 16 q rows, kv chunks of 32.
// Q,K: [B,H,S,HD] ; Vt: [B,H,HD,S] ; AO: [B,S,D] bf16.
// ---------------------------------------------------------------------------
__global__ __launch_bounds__(128)
void flash_attn_bf16(const bf16_t* __restrict__ Q, const bf16_t* __restrict__ Km,
                     const bf16_t* __restrict__ Vt, bf16_t* __restrict__ AO) {
  __shared__ alignas(16) bf16_t lP[4][16 * 32];  // per-wave P (C->A relayout)

  const int lane = threadIdx.x & 31, wave = threadIdx.x >> 5;
  const int lg = lane >> 4, ln = lane & 15;
  const int bh = blockIdx.y;
  const int b = bh >> 4, h = bh & (H_ - 1);
  const int q0 = blockIdx.x * 64 + wave * 16;

  const bf16_t* qrow = Q + ((size_t)bh * S_ + q0 + ln) * HD_;
  v16bf qf[2];
  #pragma unroll
  for (int c = 0; c < 2; ++c) {
    BF16x16 u;
    u.h[0] = *(const v8bf*)(qrow + c * 32 + lg * 16);
    u.h[1] = *(const v8bf*)(qrow + c * 32 + lg * 16 + 8);
    qf[c] = u.v;
  }

  float mrow[8], lrow[8];
  #pragma unroll
  for (int r = 0; r < 8; ++r) { mrow[r] = -3.0e38f; lrow[r] = 0.0f; }
  v8f o[4] = {};

  const float scale = 0.125f;            // 1/sqrt(64)
  const int nkc = ((q0 + 15) >> 5) + 1;  // stop at the causal diagonal
  bf16_t* myP = lP[wave];
  const bf16_t* kbh = Km + (size_t)bh * S_ * HD_;
  const bf16_t* vbh = Vt + (size_t)bh * HD_ * S_;

  for (int kc = 0; kc < nkc; ++kc) {
    const int kvb = kc * 32;
    if (kc + 1 < nkc)  // prefetch next K chunk (4KB = 32 lanes x 128B lines)
      __builtin_prefetch((const char*)(kbh + (size_t)(kvb + 32) * HD_) + lane * 128, 0, 1);

    v8f s0 = {}, s1 = {};
    #pragma unroll
    for (int c = 0; c < 2; ++c) {
      const bf16_t* kp = kbh + (size_t)kvb * HD_ + c * 32 + lg * 16;
      const v16bf k0v = *(const v16bf*)(kp + (size_t)ln * HD_);
      const v16bf k1v = *(const v16bf*)(kp + (size_t)(16 + ln) * HD_);
      s0 = wmma_bf16(qf[c], k0v, s0);
      s1 = wmma_bf16(qf[c], k1v, s1);
    }
    // scale + causal mask (C layout: row = g*8+r, col = ln)
    #pragma unroll
    for (int r = 0; r < 8; ++r) {
      const int qr = q0 + lg * 8 + r;
      float a0 = s0[r] * scale;
      float a1 = s1[r] * scale;
      if (kvb + ln > qr)      a0 = -3.0e38f;
      if (kvb + 16 + ln > qr) a1 = -3.0e38f;
      s0[r] = a0; s1[r] = a1;
    }
    float mx[8];
    #pragma unroll
    for (int r = 0; r < 8; ++r) mx[r] = fmaxf(s0[r], s1[r]);
    #pragma unroll
    for (int d = 1; d < 16; d <<= 1)
      #pragma unroll
      for (int r = 0; r < 8; ++r) mx[r] = fmaxf(mx[r], __shfl_xor(mx[r], d, 16));

    float rs[8];
    #pragma unroll
    for (int r = 0; r < 8; ++r) {
      const float mnew  = fmaxf(mrow[r], mx[r]);
      const float alpha = __expf(mrow[r] - mnew);
      mrow[r] = mnew;
      const float p0 = __expf(s0[r] - mnew);
      const float p1 = __expf(s1[r] - mnew);
      s0[r] = p0; s1[r] = p1;
      rs[r] = p0 + p1;
      lrow[r] *= alpha;
      #pragma unroll
      for (int nt = 0; nt < 4; ++nt) o[nt][r] *= alpha;
    }
    #pragma unroll
    for (int d = 1; d < 16; d <<= 1)
      #pragma unroll
      for (int r = 0; r < 8; ++r) rs[r] += __shfl_xor(rs[r], d, 16);
    #pragma unroll
    for (int r = 0; r < 8; ++r) lrow[r] += rs[r];

    // P: C-layout -> wave-private LDS -> A-layout fragment
    #pragma unroll
    for (int r = 0; r < 8; ++r) {
      myP[(lg * 8 + r) * 32 + ln]      = (bf16_t)s0[r];
      myP[(lg * 8 + r) * 32 + 16 + ln] = (bf16_t)s1[r];
    }
    BF16x16 u;
    u.h[0] = *(const v8bf*)&myP[ln * 32 + lg * 16];
    u.h[1] = *(const v8bf*)&myP[ln * 32 + lg * 16 + 8];
    const v16bf pf = u.v;

    #pragma unroll
    for (int nt = 0; nt < 4; ++nt) {
      const v16bf vf = *(const v16bf*)(vbh + (size_t)(nt * 16 + ln) * S_ + kvb + lg * 16);
      o[nt] = wmma_bf16(pf, vf, o[nt]);
    }
  }

  #pragma unroll
  for (int r = 0; r < 8; ++r) {
    const float inv = 1.0f / lrow[r];
    const int sr = q0 + lg * 8 + r;
    #pragma unroll
    for (int nt = 0; nt < 4; ++nt) {
      const int d = h * HD_ + nt * 16 + ln;
      AO[((size_t)b * S_ + sr) * D_ + d] = (bf16_t)(o[nt][r] * inv);
    }
  }
}

// ---------------------------------------------------------------------------
extern "C" void kernel_launch(void* const* d_in, const int* in_sizes, int n_in,
                              void* d_out, int out_size, void* d_ws, size_t ws_size,
                              hipStream_t stream) {
  const float* x  = (const float*)d_in[0];
  const float* Wq = (const float*)d_in[1];
  const float* Wk = (const float*)d_in[2];
  const float* Wv = (const float*)d_in[3];
  const float* Wo = (const float*)d_in[4];
  const float* bo = (const float*)d_in[5];
  float* out = (float*)d_out;

  char* ws = (char*)d_ws;
  size_t off = 0;
  auto walloc = [&](size_t bytes) -> void* {
    void* p = ws + off;
    off += (bytes + 255) & ~(size_t)255;
    return p;
  };
  bf16_t* xb   = (bf16_t*)walloc((size_t)M_ * D_ * sizeof(bf16_t));
  bf16_t* Wqt  = (bf16_t*)walloc((size_t)D_ * D_ * sizeof(bf16_t));  // [N][K]
  bf16_t* Wkt  = (bf16_t*)walloc((size_t)D_ * D_ * sizeof(bf16_t));
  bf16_t* Wvt  = (bf16_t*)walloc((size_t)D_ * D_ * sizeof(bf16_t));
  bf16_t* Wot  = (bf16_t*)walloc((size_t)D_ * D_ * sizeof(bf16_t));
  bf16_t* Qb   = (bf16_t*)walloc((size_t)M_ * D_ * sizeof(bf16_t));  // [B,H,S,HD]
  bf16_t* Kb   = (bf16_t*)walloc((size_t)M_ * D_ * sizeof(bf16_t));  // [B,H,S,HD]
  bf16_t* Vt   = (bf16_t*)walloc((size_t)M_ * D_ * sizeof(bf16_t));  // [B,H,HD,S]
  bf16_t* AO   = (bf16_t*)walloc((size_t)M_ * D_ * sizeof(bf16_t));  // [B,S,D]

  const int CT = 256 * 4;
  cvt_f32_bf16<<<(M_ * D_ + CT - 1) / CT, 256, 0, stream>>>(x, xb, M_ * D_);
  const dim3 tg(D_ / 64, D_ / 64);  // (16,16)
  cvt_transpose_f32_bf16<<<tg, 256, 0, stream>>>(Wq, Wqt, D_, D_);
  cvt_transpose_f32_bf16<<<tg, 256, 0, stream>>>(Wk, Wkt, D_, D_);
  cvt_transpose_f32_bf16<<<tg, 256, 0, stream>>>(Wv, Wvt, D_, D_);
  cvt_transpose_f32_bf16<<<tg, 256, 0, stream>>>(Wo, Wot, D_, D_);

  const dim3 gg(D_ / 128, M_ / 128);  // (8, 64)
  gemm_bf16_wmma<0><<<gg, 256, 0, stream>>>(xb, Wqt, nullptr, Qb);
  gemm_bf16_wmma<0><<<gg, 256, 0, stream>>>(xb, Wkt, nullptr, Kb);
  gemm_bf16_wmma<2><<<gg, 256, 0, stream>>>(xb, Wvt, nullptr, Vt);

  flash_attn_bf16<<<dim3(S_ / 64, B_ * H_), 128, 0, stream>>>(Qb, Kb, Vt, AO);

  gemm_bf16_wmma<1><<<gg, 256, 0, stream>>>(AO, Wot, bo, out);
}
